// TNet_LF_26783416058418
// MI455X (gfx1250) — matrix operand order repaired and checked by reading, not compiled, not run on hardware
//
#include <hip/hip_runtime.h>
#include <math.h>
#include <stdint.h>

// ---------------------------------------------------------------------------
// TNet-LF forward for gfx1250 (CDNA5). wave32, WMMA f32_16x16x32_f16 GEMMs.
// ---------------------------------------------------------------------------

typedef __attribute__((ext_vector_type(16))) _Float16 v16h;
typedef __attribute__((ext_vector_type(8)))  float    v8f;

#define B_   512
#define T_   512
#define TA_  8
#define DW_  300
#define H_   50
#define G4_  200   // 4*H
#define D2_  100   // 2*H
#define NK_  50
#define CLS_ 3

#define KP_EMB 320   // 300 padded to mult of 32
#define LDXG   208   // 200 padded to mult of 16
#define LDCAT  224   // 200 padded to mult of 32
#define LDCV   64    // 50 padded to mult of 16
#define M1_    (B_*T_)   // 262144
#define M2_    (B_*TA_)  // 4096
#define NTMAX  13        // max N tiles (208/16)
#define BSTRIDE (NTMAX*32) // halves per k-pair row in the LDS B slab

__device__ __forceinline__ float sigmf(float x) { return 1.f / (1.f + expf(-x)); }

// ---- WMMA fragment helpers (layouts per CDNA5 ISA 7.12.2) ------------------

// A (16x32 f16): lane -> m = lane%16 ; VGPR j<4: k = (lane/16)*8 + 2j(+1),
// VGPR j>=4: k += 16.
__device__ __forceinline__ v16h lds_a_frag(const _Float16* hbuf, int ldh,
                                           int kBase, int lane) {
  int m = lane & 15, half = lane >> 4;
  const _Float16* row = hbuf + m * ldh;
  v16h a;
#pragma unroll
  for (int j = 0; j < 4; ++j) {
    int k  = kBase + half * 8 + 2 * j;
    int k2 = k + 16;
    a[2 * j]     = row[k];
    a[2 * j + 1] = row[k + 1];
    a[8 + 2 * j]     = row[k2];
    a[8 + 2 * j + 1] = row[k2 + 1];
  }
  return a;
}

// B fragment for the LSTM recurrence, built straight from f32 Whh [200,50].
// B(k,n) = Whh[n][k]; zero-pad k>=50, n>=200.
__device__ __forceinline__ v16h make_whh_bfrag(const float* __restrict__ Whh,
                                               int nt, int ks, int lane) {
  int nl = lane & 15, half = lane >> 4;
  int gn = nt * 16 + nl;
  v16h b;
#pragma unroll
  for (int j = 0; j < 8; ++j) {
#pragma unroll
    for (int s = 0; s < 2; ++s) {
      int k = ks * 32 + half * 8 + 2 * (j & 3) + (j >> 2) * 16 + s;
      float v = (k < H_ && gn < G4_) ? Whh[gn * H_ + k] : 0.f;
      b[2 * j + s] = (_Float16)v;
    }
  }
  return b;
}

// ---------------------------------------------------------------------------
// WMMA GEMM, A-stationary: each wave owns one 16-row M tile, holds ONE A
// fragment per K-step and sweeps ALL N tiles (NT accumulators, compile-time)
// against a double-buffered 32xN f16 B slab in LDS. A read exactly once from
// HBM; no per-tile branches so DS loads pipeline ahead of the WMMA chain.
//   BMODE 0: B(k,n) = Bw[n*Kvalid + k]        (weights stored [N,K], f32)
//   BMODE 1: B(k,n) = Bw[n*300 + (k%100)*3 + k/100]   (conv_w [NK,2H,3])
//   EPI 0: Cf32[m*ldc+n] = acc+bias
//   EPI 1: Cf16[m*ldc+n] = (f16)(acc+bias)
//   EPI 2: hs[m*100+n] = (relu(acc+bias) + hs[m*100+n]) * pos[m]   (in place)
// grid = (1, M/128), block = 256 (8 waves, one M tile each)
// ---------------------------------------------------------------------------
template <int NT, int BMODE, int EPI>
__global__ __launch_bounds__(256) void gemm16_wmma(
    const _Float16* __restrict__ A, int lda,
    const float* __restrict__ Bw, int Kvalid, int Nvalid, int Ksteps,
    const float* __restrict__ bias1, const float* __restrict__ bias2,
    float* Cf32, _Float16* Cf16, int ldc,
    const float* __restrict__ pos) {
  // pair-swizzled slab: [(k&31)/2]*BSTRIDE + nt*32 + 2*n + (k&1)
  __shared__ _Float16 Bs[2][16 * BSTRIDE];

  const int lane = threadIdx.x & 31;
  const int nl = lane & 15, half = lane >> 4;
  const int tm = blockIdx.y * 8 + (threadIdx.x >> 5);
  const _Float16* Arow = A + (size_t)(tm * 16 + nl) * lda;  // A: m = lane%16

  auto stage = [&](int ks, int buf) {
    const int total = NT * 16 * 32;
    for (int idx = threadIdx.x; idx < total; idx += 256) {
      int col = idx >> 5, kk = idx & 31;     // col = global n (grid.x == 1)
      int k = ks * 32 + kk;
      float v = 0.f;
      if (k < Kvalid && col < Nvalid) {
        v = (BMODE == 0) ? Bw[col * Kvalid + k]
                         : Bw[col * 300 + (k % 100) * 3 + (k / 100)];
      }
      Bs[buf][(kk >> 1) * BSTRIDE + (col >> 4) * 32 + 2 * (col & 15) + (kk & 1)] =
          (_Float16)v;
    }
  };

  v8f acc[NT];
#pragma unroll
  for (int nt = 0; nt < NT; ++nt) acc[nt] = (v8f){};

  stage(0, 0);
  __syncthreads();
  int buf = 0;
  for (int ks = 0; ks < Ksteps; ++ks) {
    if (ks + 1 < Ksteps) {
      stage(ks + 1, buf ^ 1);                          // overlaps WMMAs below
      __builtin_prefetch(Arow + (ks + 1) * 32, 0, 3);  // next A fragment
    }
    v16h a;
#pragma unroll
    for (int j = 0; j < 4; ++j) {
      int k = ks * 32 + half * 8 + 2 * j;
      a[2 * j]         = Arow[k];
      a[2 * j + 1]     = Arow[k + 1];
      a[8 + 2 * j]     = Arow[k + 16];
      a[8 + 2 * j + 1] = Arow[k + 17];
    }
#pragma unroll
    for (int nt = 0; nt < NT; ++nt) {
      v16h b;
#pragma unroll
      for (int j = 0; j < 8; ++j) {
        int kk = half * 8 + 2 * (j & 3) + (j >> 2) * 16;
        b[2 * j]     = Bs[buf][(kk >> 1) * BSTRIDE + nt * 32 + 2 * nl];
        b[2 * j + 1] = Bs[buf][(kk >> 1) * BSTRIDE + nt * 32 + 2 * nl + 1];
      }
      acc[nt] = __builtin_amdgcn_wmma_f32_16x16x32_f16(false, a, false, b,
                                                       (short)0, acc[nt],
                                                       false, false);
    }
    __syncthreads();
    buf ^= 1;
  }

#pragma unroll
  for (int nt = 0; nt < NT; ++nt) {
    int gn = nt * 16 + nl;
    if (gn < Nvalid) {
      float bs = (bias1 ? bias1[gn] : 0.f) + (bias2 ? bias2[gn] : 0.f);
#pragma unroll
      for (int r = 0; r < 8; ++r) {
        int m = tm * 16 + r + 8 * half;  // C/D: m = r + 8*(lane/16)
        float v = acc[nt][r] + bs;
        if (EPI == 1) {
          Cf16[(size_t)m * ldc + gn] = (_Float16)v;
        } else if (EPI == 2) {
          float h0 = Cf32[(size_t)m * D2_ + gn];
          v = fmaxf(v, 0.f);
          Cf32[(size_t)m * D2_ + gn] = (v + h0) * pos[m];
        } else {
          Cf32[(size_t)m * ldc + gn] = v;
        }
      }
    }
  }
}

// ---------------------------------------------------------------------------
// LSTM recurrence. One workgroup = 16 batch rows, 4 waves. Whh B-fragments
// live in VGPRs for the whole T loop; h (f16) in LDS; gates staged via LDS.
// xg [B*T, LDXG] f16 (bih+bhh folded in) is software-pipelined: step s issues
// the dword loads for step s+1 into registers BEFORE the WMMA phase, dumps
// them to a double-buffered LDS slab after it, and the elementwise update
// adds xg from LDS (so WMMA C starts from inline zero).
// hsOut: [B*T, 100] f32.
// ---------------------------------------------------------------------------
__global__ __launch_bounds__(128) void lstm_rec(
    const _Float16* __restrict__ xg, const float* __restrict__ Whh,
    float* __restrict__ hsOut, int T, int colOff, int rev) {
  __shared__ _Float16 h16s[16 * 64];
  __shared__ _Float16 xgS[2][16 * LDXG];
  __shared__ float Gs[16 * LDXG];

  const int tid = threadIdx.x;
  const int wave = tid >> 5, lane = tid & 31;
  const int nl = lane & 15, half = lane >> 4;
  const int b0 = blockIdx.x * 16;

  for (int i = tid; i < 16 * 64; i += 128) h16s[i] = (_Float16)0.f;

  v16h bf[4][2];
#pragma unroll
  for (int j = 0; j < 4; ++j) {
    int nt = wave + 4 * j;
    if (nt < 13) {
      bf[j][0] = make_whh_bfrag(Whh, nt, 0, lane);
      bf[j][1] = make_whh_bfrag(Whh, nt, 1, lane);
    }
  }
  float cst[7];
#pragma unroll
  for (int j = 0; j < 7; ++j) cst[j] = 0.f;

  // prologue: stage xg(t0) into slab 0.  16 rows x 104 dwords = 13 dw/thread
  {
    int t0 = rev ? (T - 1) : 0;
#pragma unroll
    for (int i = 0; i < 13; ++i) {
      int idx = tid + 128 * i;
      int row = idx / 104, d2 = idx % 104;
      *(uint32_t*)&xgS[0][row * LDXG + 2 * d2] =
          *(const uint32_t*)(xg + ((size_t)(b0 + row) * T + t0) * LDXG + 2 * d2);
    }
  }
  __syncthreads();

  int buf = 0;
  for (int s = 0; s < T; ++s) {
    const int t = rev ? (T - 1 - s) : s;
    const int havenext = (s + 1 < T);
    const int tn = rev ? (T - 2 - s) : (s + 1);

    // issue next step's xg loads; they fly while the WMMAs below execute
    uint32_t ld[13];
    if (havenext) {
#pragma unroll
      for (int i = 0; i < 13; ++i) {
        int idx = tid + 128 * i;
        int row = idx / 104, d2 = idx % 104;
        ld[i] = *(const uint32_t*)(xg +
                 ((size_t)(b0 + row) * T + tn) * LDXG + 2 * d2);
      }
    }

    v16h a0 = lds_a_frag(h16s, 64, 0, lane);
    v16h a1 = lds_a_frag(h16s, 64, 32, lane);
#pragma unroll
    for (int j = 0; j < 4; ++j) {
      int nt = wave + 4 * j;       // wave-uniform branch: EXEC stays full
      if (nt < 13) {
        int gn = nt * 16 + nl;
        v8f acc = (v8f){};         // C = 0; xg added in the update phase
        acc = __builtin_amdgcn_wmma_f32_16x16x32_f16(false, a0, false, bf[j][0],
                                                     (short)0, acc, false, false);
        acc = __builtin_amdgcn_wmma_f32_16x16x32_f16(false, a1, false, bf[j][1],
                                                     (short)0, acc, false, false);
#pragma unroll
        for (int r = 0; r < 8; ++r)
          Gs[(r + 8 * half) * LDXG + gn] = acc[r];
      }
    }

    // drain the in-flight loads into the other slab
    if (havenext) {
#pragma unroll
      for (int i = 0; i < 13; ++i) {
        int idx = tid + 128 * i;
        int row = idx / 104, d2 = idx % 104;
        *(uint32_t*)&xgS[buf ^ 1][row * LDXG + 2 * d2] = ld[i];
      }
    }
    __syncthreads();

    // elementwise gate update: 16*50 = 800 (b,h) pairs over 128 threads
    const _Float16* xgc = xgS[buf];
#pragma unroll
    for (int j = 0; j < 7; ++j) {
      int p = tid + 128 * j;
      if (p < 800) {
        int bl = p / 50, h = p % 50;
        float gi = Gs[bl * LDXG + h]        + (float)xgc[bl * LDXG + h];
        float gf = Gs[bl * LDXG + 50 + h]   + (float)xgc[bl * LDXG + 50 + h];
        float gg = Gs[bl * LDXG + 100 + h]  + (float)xgc[bl * LDXG + 100 + h];
        float go = Gs[bl * LDXG + 150 + h]  + (float)xgc[bl * LDXG + 150 + h];
        float c = sigmf(gf) * cst[j] + sigmf(gi) * tanhf(gg);
        cst[j] = c;
        float hv = sigmf(go) * tanhf(c);
        h16s[bl * 64 + h] = (_Float16)hv;
        hsOut[((size_t)(b0 + bl) * T + t) * D2_ + colOff + h] = hv;
      }
    }
    __syncthreads();
    buf ^= 1;
  }
}

// ---------------------------------------------------------------------------
// Attention (t-dim = 8): fused scores + softmax + context; stages [hs,ri]
// as f16 into cat16 [B*T, 224] (pad zeroed).
// ---------------------------------------------------------------------------
__global__ __launch_bounds__(128) void attn_stage(
    const float* __restrict__ hs, const float* __restrict__ ha,
    _Float16* __restrict__ cat16) {
  __shared__ float haS[TA_ * D2_];
  const int b = blockIdx.y, tid = threadIdx.x;
  for (int i = tid; i < TA_ * D2_; i += 128) haS[i] = ha[(size_t)b * TA_ * D2_ + i];
  __syncthreads();

  const int sI = blockIdx.x * 128 + tid;
  const float* hrow = hs + ((size_t)b * T_ + sI) * D2_;
  float sc[TA_];
#pragma unroll
  for (int t = 0; t < TA_; ++t) sc[t] = 0.f;
  for (int d = 0; d < D2_; ++d) {
    float x = hrow[d];
#pragma unroll
    for (int t = 0; t < TA_; ++t) sc[t] += x * haS[t * D2_ + d];
  }
  float mx = sc[0];
#pragma unroll
  for (int t = 1; t < TA_; ++t) mx = fmaxf(mx, sc[t]);
  float sum = 0.f;
#pragma unroll
  for (int t = 0; t < TA_; ++t) { sc[t] = expf(sc[t] - mx); sum += sc[t]; }
  float inv = 1.f / sum;

  _Float16* crow = cat16 + ((size_t)b * T_ + sI) * LDCAT;
  for (int d = 0; d < D2_; ++d) {
    float r = 0.f;
#pragma unroll
    for (int t = 0; t < TA_; ++t) r += sc[t] * haS[t * D2_ + d];
    crow[d]        = (_Float16)hrow[d];
    crow[D2_ + d]  = (_Float16)(r * inv);
  }
#pragma unroll
  for (int d = 2 * D2_; d < LDCAT; ++d) crow[d] = (_Float16)0.f;
}

// ---- small utility kernels -------------------------------------------------

__global__ void embed_pack(const float* __restrict__ emb,
                           const int* __restrict__ tok,
                           _Float16* __restrict__ out, int rows) {
  int tid = blockIdx.x * 256 + threadIdx.x;
  int total = rows * (KP_EMB / 4);
  if (tid >= total) return;
  int row = tid / (KP_EMB / 4), g = tid % (KP_EMB / 4);
  const float* e = emb + (size_t)tok[row] * DW_;
#pragma unroll
  for (int j = 0; j < 4; ++j) {
    int k = g * 4 + j;
    out[(size_t)row * KP_EMB + k] = (k < DW_) ? (_Float16)e[k] : (_Float16)0.f;
  }
}

__global__ void im2col_pack(const float* __restrict__ hs,
                            _Float16* __restrict__ col) {
  int tid = blockIdx.x * 256 + threadIdx.x;
  int total = M1_ * (KP_EMB / 4);
  if (tid >= total) return;
  int m = tid / (KP_EMB / 4), g = tid % (KP_EMB / 4);
  int b = m >> 9, t = m & 511;
#pragma unroll
  for (int j = 0; j < 4; ++j) {
    int k = g * 4 + j;
    float v = 0.f;
    if (k < 300) {
      int tap = k / 100, ch = k % 100;
      int tt = t + tap - 1;
      if (tt >= 0 && tt < T_) v = hs[((size_t)b * T_ + tt) * D2_ + ch];
    }
    col[(size_t)m * KP_EMB + k] = (_Float16)v;
  }
}

__global__ void lengths_k(const int* __restrict__ sents,
                          const int* __restrict__ aspect,
                          int* slen, int* alen) {
  int b = blockIdx.x * 64 + threadIdx.x;
  if (b >= B_) return;
  int c = 0;
  for (int t = 0; t < T_; ++t) c += (sents[b * T_ + t] != 0);
  slen[b] = c;
  c = 0;
  for (int t = 0; t < TA_; ++t) c += (aspect[b * TA_ + t] != 0);
  alen[b] = c;
}

__global__ void pos_k(const int* __restrict__ aIdx, const int* __restrict__ alen,
                      const int* __restrict__ slen, float* __restrict__ pos) {
  int tid = blockIdx.x * 256 + threadIdx.x;
  if (tid >= M1_) return;
  int b = tid >> 9, t = tid & 511;
  int aI = aIdx[b];
  int aend = aI + alen[b];
  float p;
  if (t < aend)          p = 1.f - ((float)aend - (float)t) / 40.f;
  else if (t < slen[b])  p = 1.f - ((float)t - (float)aI) / 40.f;
  else                   p = 0.f;
  pos[tid] = p;
}

__global__ void maxpool_k(const float* __restrict__ conv,
                          const float* __restrict__ convb,
                          float* __restrict__ pooled) {
  int tid = blockIdx.x * 64 + threadIdx.x;
  int b = tid >> 6, n = tid & 63;
  if (b >= B_ || n >= NK_) return;
  float m = -3.4e38f;
  const float* base = conv + (size_t)b * T_ * LDCV + n;
  for (int t = 0; t < T_; ++t) m = fmaxf(m, base[(size_t)t * LDCV]);
  pooled[b * LDCV + n] = m + convb[n];
}

__global__ void pred_k(const float* __restrict__ pooled,
                       const float* __restrict__ pw,
                       const float* __restrict__ pb, float* __restrict__ out) {
  int tid = blockIdx.x * 64 + threadIdx.x;
  if (tid >= B_ * CLS_) return;
  int b = tid / CLS_, c = tid % CLS_;
  float s = pb[c];
  for (int n = 0; n < NK_; ++n) s += pooled[b * LDCV + n] * pw[c * NK_ + n];
  out[tid] = s;
}

// ---------------------------------------------------------------------------
extern "C" void kernel_launch(void* const* d_in, const int* in_sizes, int n_in,
                              void* d_out, int out_size, void* d_ws, size_t ws_size,
                              hipStream_t stream) {
  (void)in_sizes; (void)n_in; (void)out_size; (void)ws_size;
  // params in setup_inputs() insertion order
  const float* emb      = (const float*)d_in[0];
  const float* Wih[4]   = {(const float*)d_in[1],  (const float*)d_in[5],
                           (const float*)d_in[9],  (const float*)d_in[13]};
  const float* Whh[4]   = {(const float*)d_in[2],  (const float*)d_in[6],
                           (const float*)d_in[10], (const float*)d_in[14]};
  const float* bih[4]   = {(const float*)d_in[3],  (const float*)d_in[7],
                           (const float*)d_in[11], (const float*)d_in[15]};
  const float* bhh[4]   = {(const float*)d_in[4],  (const float*)d_in[8],
                           (const float*)d_in[12], (const float*)d_in[16]};
  const float* Wt_w     = (const float*)d_in[17];
  const float* Wt_b     = (const float*)d_in[18];
  const float* conv_w   = (const float*)d_in[19];
  const float* conv_b   = (const float*)d_in[20];
  const float* pred_w   = (const float*)d_in[21];
  const float* pred_b   = (const float*)d_in[22];
  const int*   sents    = (const int*)d_in[23];
  const int*   aspect   = (const int*)d_in[24];
  const int*   aspectI  = (const int*)d_in[25];
  float* out = (float*)d_out;

  // workspace carve-up (256B aligned)
  auto align = [](size_t x) { return (x + 255) & ~(size_t)255; };
  char* p = (char*)d_ws;
  size_t o = 0;
  _Float16* xs16  = (_Float16*)(p + o); o = align(o + (size_t)M1_ * KP_EMB * 2);
  _Float16* xa16  = (_Float16*)(p + o); o = align(o + (size_t)M2_ * KP_EMB * 2);
  _Float16* xg1f  = (_Float16*)(p + o); size_t xg1f_off = o;
                                        o = align(o + (size_t)M1_ * LDXG * 2);
  _Float16* xg1b  = (_Float16*)(p + o); o = align(o + (size_t)M1_ * LDXG * 2);
  _Float16* xg2f  = (_Float16*)(p + o); o = align(o + (size_t)M2_ * LDXG * 2);
  _Float16* xg2b  = (_Float16*)(p + o); o = align(o + (size_t)M2_ * LDXG * 2);
  float*    hs    = (float*)(p + o);    o = align(o + (size_t)M1_ * D2_ * 4);
  float*    ha    = (float*)(p + o);    o = align(o + (size_t)M2_ * D2_ * 4);
  _Float16* cat16 = (_Float16*)(p + o); o = align(o + (size_t)M1_ * LDCAT * 2);
  float*    pos   = (float*)(p + o);    o = align(o + (size_t)M1_ * 4);
  int*      slen  = (int*)(p + o);      o = align(o + B_ * 4);
  int*      alen  = (int*)(p + o);      o = align(o + B_ * 4);
  float*    pooled= (float*)(p + o);    o = align(o + B_ * LDCV * 4);
  float*    convo = (float*)(p + xg1f_off);  // reuse xg1f after LSTM (67MB<109MB)

  // 1) lengths + position-decay mask
  lengths_k<<<dim3((B_ + 63) / 64), dim3(64), 0, stream>>>(sents, aspect, slen, alen);
  pos_k<<<dim3((M1_ + 255) / 256), dim3(256), 0, stream>>>(aspectI, alen, slen, pos);

  // 2) embedding gather -> f16 (K padded to 320, zero-filled)
  embed_pack<<<dim3((M1_ * (KP_EMB / 4) + 255) / 256), dim3(256), 0, stream>>>(emb, sents, xs16, M1_);
  embed_pack<<<dim3((M2_ * (KP_EMB / 4) + 255) / 256), dim3(256), 0, stream>>>(emb, aspect, xa16, M2_);

  // 3) input projections xg = x @ Wih^T + bih + bhh  (WMMA, f16 out)
  //    A read exactly once: all 13 N tiles per wave.
  {
    dim3 blk(256);
    dim3 g1(1, M1_ / 128), g2(1, M2_ / 128);
    gemm16_wmma<13, 0, 1><<<g1, blk, 0, stream>>>(xs16, KP_EMB, Wih[0], DW_, G4_,
                                                  KP_EMB / 32, bih[0], bhh[0],
                                                  nullptr, xg1f, LDXG, nullptr);
    gemm16_wmma<13, 0, 1><<<g1, blk, 0, stream>>>(xs16, KP_EMB, Wih[1], DW_, G4_,
                                                  KP_EMB / 32, bih[1], bhh[1],
                                                  nullptr, xg1b, LDXG, nullptr);
    gemm16_wmma<13, 0, 1><<<g2, blk, 0, stream>>>(xa16, KP_EMB, Wih[2], DW_, G4_,
                                                  KP_EMB / 32, bih[2], bhh[2],
                                                  nullptr, xg2f, LDXG, nullptr);
    gemm16_wmma<13, 0, 1><<<g2, blk, 0, stream>>>(xa16, KP_EMB, Wih[3], DW_, G4_,
                                                  KP_EMB / 32, bih[3], bhh[3],
                                                  nullptr, xg2b, LDXG, nullptr);
  }

  // 4) BiLSTM recurrences (weight-stationary WMMA, pipelined xg through LDS)
  lstm_rec<<<dim3(B_ / 16), dim3(128), 0, stream>>>(xg1f, Whh[0], hs, T_,  0, 0);
  lstm_rec<<<dim3(B_ / 16), dim3(128), 0, stream>>>(xg1b, Whh[1], hs, T_,  H_, 1);
  lstm_rec<<<dim3(B_ / 16), dim3(128), 0, stream>>>(xg2f, Whh[2], ha, TA_, 0, 0);
  lstm_rec<<<dim3(B_ / 16), dim3(128), 0, stream>>>(xg2b, Whh[3], ha, TA_, H_, 1);

  // 5) L=2 CPT layers: attention -> Wt GEMM (fused relu + residual + pos)
  for (int it = 0; it < 2; ++it) {
    attn_stage<<<dim3(T_ / 128, B_), dim3(128), 0, stream>>>(hs, ha, cat16);
    gemm16_wmma<7, 0, 2><<<dim3(1, M1_ / 128), dim3(256), 0, stream>>>(
        cat16, LDCAT, Wt_w, G4_, D2_, LDCAT / 32,
        Wt_b, nullptr, hs, nullptr, D2_, pos);
  }

  // 6) conv1d(k=3, pad 1) as im2col GEMM (reuse xs16 as the im2col buffer)
  im2col_pack<<<dim3((M1_ * (KP_EMB / 4) + 255) / 256), dim3(256), 0, stream>>>(hs, xs16);
  gemm16_wmma<4, 1, 0><<<dim3(1, M1_ / 128), dim3(256), 0, stream>>>(
      xs16, KP_EMB, conv_w, DW_, NK_, KP_EMB / 32,
      nullptr, nullptr, convo, nullptr, LDCV, nullptr);

  // 7) max-pool over T + classifier head
  maxpool_k<<<dim3(B_ * LDCV / 64), dim3(64), 0, stream>>>(convo, conv_b, pooled);
  pred_k<<<dim3((B_ * CLS_ + 63) / 64), dim3(64), 0, stream>>>(pooled, pred_w, pred_b, out);
}
